// Attention_7421703487529
// MI455X (gfx1250) — compile-verified, matrix operand order checked by reading, not run on hardware
//
#include <hip/hip_runtime.h>

// ---------------------------------------------------------------------------
// MHA forward for B=4, N=2048, C=512, H=8, D=64 on gfx1250 (wave32, WMMA).
// Pipeline: one-time fp32->bf16 convert -> QKV bf16-WMMA GEMM ->
//           flash attention (async-LDS staged K/V, double buffered) ->
//           proj GEMM (fp32 out).
// ---------------------------------------------------------------------------

typedef __attribute__((ext_vector_type(16))) __bf16 v16bf;
typedef __attribute__((ext_vector_type(8)))  float  v8f;

#define DIM    512
#define NSEQ   2048
#define BATCH  4
#define HEADS  8
#define HD     64
#define SCALE  0.125f                  // 64^-0.5
#define SC2    (0.125f * 1.44269504088896340736f)   // SCALE * log2(e)
#define QKVN   (3 * DIM)

// One WMMA operand fragment: 16 bf16 per lane (8 VGPRs).
union ABfrag {
  v16bf          v;
  unsigned short u[16];
  uint4          q4[2];
};

__device__ __forceinline__ unsigned short f2bf(float f) {
  __bf16 h = (__bf16)f;
  return __builtin_bit_cast(unsigned short, h);
}

__device__ __forceinline__ v8f wmma_bf16(const ABfrag& a, const ABfrag& b, v8f c) {
  return __builtin_amdgcn_wmma_f32_16x16x32_bf16(false, a.v, false, b.v,
                                                 (short)0, c, false, false);
}

// CDNA5 async global->LDS copy, 16B per lane, tracked by ASYNCcnt.
__device__ __forceinline__ void async_b128(unsigned lds_byte, unsigned g_byte,
                                           const void* sbase) {
  asm volatile("global_load_async_to_lds_b128 %0, %1, %2"
               :: "v"(lds_byte), "v"(g_byte), "s"(sbase) : "memory");
}
__device__ __forceinline__ void wait_async0() {
  asm volatile("s_wait_asynccnt 0x0" ::: "memory");
}

// ---------------------------------------------------------------------------
// Kernel 0: one-time fp32 -> bf16 convert (float4 -> ushort4).
// ---------------------------------------------------------------------------
__global__ __launch_bounds__(256) void cvt_bf16_kernel(
    const float* __restrict__ in, unsigned short* __restrict__ out, int n4) {
  const int i = blockIdx.x * blockDim.x + threadIdx.x;
  if (i < n4) {
    const float4 f = ((const float4*)in)[i];
    ushort4 h;
    h.x = f2bf(f.x); h.y = f2bf(f.y); h.z = f2bf(f.z); h.w = f2bf(f.w);
    ((ushort4*)out)[i] = h;
  }
}

// ---------------------------------------------------------------------------
// Kernel 1: Y = X @ Wqkv^T  (bf16 WMMA, fp32 accum) ->
//   q,k : [B,H,N,64] bf16      vT : [B,H,64,N] bf16
// One wave = 32(M) x 64(N) tile; hot loop = 12 b128 loads + 8 wmma / K-chunk.
// ---------------------------------------------------------------------------
__global__ __launch_bounds__(32) void qkv_gemm_kernel(
    const unsigned short* __restrict__ x, const unsigned short* __restrict__ w,
    unsigned short* __restrict__ q, unsigned short* __restrict__ k,
    unsigned short* __restrict__ vT) {
  const int lane = threadIdx.x;
  const int lm   = lane & 15;
  const int half = lane >> 4;
  const int klo0 = half * 8;
  const int m0   = blockIdx.x * 32;
  const int n0   = blockIdx.y * 64;

  v8f acc[2][4];
#pragma unroll
  for (int mi = 0; mi < 2; ++mi)
#pragma unroll
    for (int ni = 0; ni < 4; ++ni) acc[mi][ni] = (v8f){};

  for (int kc = 0; kc < DIM; kc += 32) {
    ABfrag af[2];
#pragma unroll
    for (int mi = 0; mi < 2; ++mi) {
      const unsigned short* ap =
          x + (size_t)(m0 + mi * 16 + lm) * DIM + kc + klo0;
      af[mi].q4[0] = *(const uint4*)(ap);
      af[mi].q4[1] = *(const uint4*)(ap + 16);
    }
    ABfrag bf[4];
#pragma unroll
    for (int ni = 0; ni < 4; ++ni) {
      const unsigned short* bp =
          w + (size_t)(n0 + ni * 16 + lm) * DIM + kc + klo0;
      bf[ni].q4[0] = *(const uint4*)(bp);
      bf[ni].q4[1] = *(const uint4*)(bp + 16);
    }
#pragma unroll
    for (int mi = 0; mi < 2; ++mi)
#pragma unroll
      for (int ni = 0; ni < 4; ++ni)
        acc[mi][ni] = wmma_bf16(af[mi], bf[ni], acc[mi][ni]);
  }

  const int which = n0 / DIM;
  const int c0    = n0 % DIM;
  const int h     = c0 / HD;
  const int bb    = m0 / NSEQ;
  const int ns0   = m0 % NSEQ;
#pragma unroll
  for (int mi = 0; mi < 2; ++mi)
#pragma unroll
    for (int ni = 0; ni < 4; ++ni) {
      const int d = ni * 16 + lm;
      if (which == 2) {
        uint4 pk;
        pk.x = (unsigned)f2bf(acc[mi][ni][0]) | ((unsigned)f2bf(acc[mi][ni][1]) << 16);
        pk.y = (unsigned)f2bf(acc[mi][ni][2]) | ((unsigned)f2bf(acc[mi][ni][3]) << 16);
        pk.z = (unsigned)f2bf(acc[mi][ni][4]) | ((unsigned)f2bf(acc[mi][ni][5]) << 16);
        pk.w = (unsigned)f2bf(acc[mi][ni][6]) | ((unsigned)f2bf(acc[mi][ni][7]) << 16);
        const int nseq = ns0 + mi * 16 + half * 8;
        *(uint4*)&vT[((size_t)(bb * HEADS + h) * HD + d) * NSEQ + nseq] = pk;
      } else {
        unsigned short* dst = (which == 0) ? q : k;
#pragma unroll
        for (int r = 0; r < 8; ++r) {
          const int nseq = ns0 + mi * 16 + half * 8 + r;
          dst[((size_t)(bb * HEADS + h) * NSEQ + nseq) * HD + d] =
              f2bf(acc[mi][ni][r]);
        }
      }
    }
}

// ---------------------------------------------------------------------------
// Kernel 2: flash attention. 4 waves/WG share one (b,h); wave w owns queries
// [m0+16w, m0+16w+16). Per 64-key step: K tile (64x64) and V^T tile (64x64)
// are staged into LDS with global_load_async_to_lds_b128 (double-buffered,
// prefetch of step i+1 overlaps compute of step i), then each wave does
// 8 WMMAs for S = Q K^T, online softmax in exp2 domain, and 8 WMMAs for
// O += P V. One s_wait_asynccnt + one barrier per step.
// ---------------------------------------------------------------------------
__global__ __launch_bounds__(128) void flash_attn_kernel(
    const unsigned short* __restrict__ q, const unsigned short* __restrict__ k,
    const unsigned short* __restrict__ vT, unsigned short* __restrict__ attn) {
  __shared__ __align__(16) unsigned short kv[2][2][64 * 64];   // 32 KB
  __shared__ __align__(16) unsigned short pstage[4][16 * 72];  //  9 KB

  const int tid  = threadIdx.x;        // 0..127
  const int wave = tid >> 5;
  const int lane = tid & 31;
  const int lm   = lane & 15;
  const int half = lane >> 4;
  const int klo0 = half * 8;
  const int m0   = blockIdx.x * 64 + wave * 16;
  const int h    = blockIdx.y;
  const int b    = blockIdx.z;

  const unsigned short* qb = q  + (size_t)(b * HEADS + h) * NSEQ * HD;
  const unsigned short* kb = k  + (size_t)(b * HEADS + h) * NSEQ * HD;
  const unsigned short* vb = vT + (size_t)(b * HEADS + h) * HD * NSEQ;

  // Cooperative async stage of one 64-key K tile + V^T tile into kv[buf].
  auto stage = [&](int buf, int kb0) {
#pragma unroll
    for (int j = 0; j < 4; ++j) {                 // K: contiguous 8 KB
      const int idx = j * 1024 + tid * 8;
      async_b128((unsigned)(uintptr_t)&kv[buf][0][idx],
                 (unsigned)((kb0 * HD + idx) * 2), kb);
    }
#pragma unroll
    for (int j = 0; j < 4; ++j) {                 // V^T: 64 rows x 128 B
      const int e = j * 128 + tid;
      const int row = e >> 3, col = (e & 7) * 8;
      async_b128((unsigned)(uintptr_t)&kv[buf][1][row * HD + col],
                 (unsigned)((row * NSEQ + kb0 + col) * 2), vb);
    }
  };

  // Q fragments (resident for the whole loop).
  ABfrag qf[2];
#pragma unroll
  for (int c = 0; c < 2; ++c) {
    const unsigned short* p = qb + (size_t)(m0 + lm) * HD + c * 32 + klo0;
    qf[c].q4[0] = *(const uint4*)(p);
    qf[c].q4[1] = *(const uint4*)(p + 16);
  }

  v8f o[4];
#pragma unroll
  for (int t = 0; t < 4; ++t) o[t] = (v8f){};
  float m_i[8], l_i[8];
#pragma unroll
  for (int r = 0; r < 8; ++r) { m_i[r] = -1e30f; l_i[r] = 0.0f; }

  stage(0, 0);                                    // prologue prefetch

  for (int it = 0; it < NSEQ / 64; ++it) {
    const int buf = it & 1;
    wait_async0();
    __syncthreads();                 // tile[buf] ready; all waves past reads
    if (it + 1 < NSEQ / 64) stage(buf ^ 1, (it + 1) * 64);

    // S = Q K^T : 4 key sub-tiles x 2 d-chunks = 8 WMMAs.
    v8f s[4];
#pragma unroll
    for (int st = 0; st < 4; ++st) {
      s[st] = (v8f){};
#pragma unroll
      for (int c = 0; c < 2; ++c) {
        ABfrag kf;
        const unsigned short* p = &kv[buf][0][(st * 16 + lm) * HD + c * 32 + klo0];
        kf.q4[0] = *(const uint4*)(p);
        kf.q4[1] = *(const uint4*)(p + 16);
        s[st] = wmma_bf16(qf[c], kf, s[st]);
      }
    }

    // Online softmax over 64 keys (exp2 domain; logits pre-scaled by SC2).
    unsigned short* pw = &pstage[wave][0];
    float alpha[8];
#pragma unroll
    for (int r = 0; r < 8; ++r) {
      float a0 = s[0][r] * SC2, a1 = s[1][r] * SC2;
      float a2 = s[2][r] * SC2, a3 = s[3][r] * SC2;
      float mx = fmaxf(fmaxf(a0, a1), fmaxf(a2, a3));
#pragma unroll
      for (int d = 8; d >= 1; d >>= 1) mx = fmaxf(mx, __shfl_xor(mx, d, 32));
      const float mnew = fmaxf(m_i[r], mx);
      const float al   = exp2f(m_i[r] - mnew);
      const float p0 = exp2f(a0 - mnew), p1 = exp2f(a1 - mnew);
      const float p2 = exp2f(a2 - mnew), p3 = exp2f(a3 - mnew);
      float rs = (p0 + p1) + (p2 + p3);
#pragma unroll
      for (int d = 8; d >= 1; d >>= 1) rs += __shfl_xor(rs, d, 32);
      l_i[r]   = l_i[r] * al + rs;
      m_i[r]   = mnew;
      alpha[r] = al;
      const int row = (half * 8 + r) * 72;        // 144B row stride, 16B mult
      pw[row + lm]      = f2bf(p0);
      pw[row + lm + 16] = f2bf(p1);
      pw[row + lm + 32] = f2bf(p2);
      pw[row + lm + 48] = f2bf(p3);
    }
    asm volatile("" ::: "memory");   // keep ds_loads after ds_stores
    ABfrag pf[2];
#pragma unroll
    for (int kh = 0; kh < 2; ++kh) {
      const unsigned short* pp = pw + lm * 72 + kh * 32 + klo0;
      pf[kh].q4[0] = *(const uint4*)(pp);
      pf[kh].q4[1] = *(const uint4*)(pp + 16);
    }
    asm volatile("" ::: "memory");

    // O = O*alpha + P V : 4 d-tiles x 2 key-halves = 8 WMMAs.
#pragma unroll
    for (int t = 0; t < 4; ++t) {
#pragma unroll
      for (int r = 0; r < 8; ++r) o[t][r] *= alpha[r];
#pragma unroll
      for (int kh = 0; kh < 2; ++kh) {
        ABfrag vf;
        const unsigned short* vp =
            &kv[buf][1][(t * 16 + lm) * HD + kh * 32 + klo0];
        vf.q4[0] = *(const uint4*)(vp);
        vf.q4[1] = *(const uint4*)(vp + 16);
        o[t] = wmma_bf16(pf[kh], vf, o[t]);
      }
    }
  }

  // Epilogue: normalize, write bf16 into [B, N, C].
#pragma unroll
  for (int r = 0; r < 8; ++r) l_i[r] = 1.0f / l_i[r];
#pragma unroll
  for (int t = 0; t < 4; ++t) {
    const int d = t * 16 + lm;
#pragma unroll
    for (int r = 0; r < 8; ++r) {
      const int m = m0 + half * 8 + r;
      attn[((size_t)(b * NSEQ + m)) * DIM + h * HD + d] =
          f2bf(o[t][r] * l_i[r]);
    }
  }
}

// ---------------------------------------------------------------------------
// Kernel 3: out = attn_bf16 @ Wproj_bf16^T + bias, fp32 out. 32x32 per wave.
// ---------------------------------------------------------------------------
__global__ __launch_bounds__(32) void proj_gemm_kernel(
    const unsigned short* __restrict__ a, const unsigned short* __restrict__ w,
    const float* __restrict__ bias, float* __restrict__ out) {
  const int lane = threadIdx.x;
  const int lm   = lane & 15;
  const int half = lane >> 4;
  const int klo0 = half * 8;
  const int m0   = blockIdx.x * 32;
  const int n0   = blockIdx.y * 32;

  v8f acc[2][2];
#pragma unroll
  for (int mi = 0; mi < 2; ++mi)
#pragma unroll
    for (int ni = 0; ni < 2; ++ni) acc[mi][ni] = (v8f){};

  for (int kc = 0; kc < DIM; kc += 32) {
    ABfrag af[2];
#pragma unroll
    for (int mi = 0; mi < 2; ++mi) {
      const unsigned short* p =
          a + (size_t)(m0 + mi * 16 + lm) * DIM + kc + klo0;
      af[mi].q4[0] = *(const uint4*)(p);
      af[mi].q4[1] = *(const uint4*)(p + 16);
    }
    ABfrag bf[2];
#pragma unroll
    for (int ni = 0; ni < 2; ++ni) {
      const unsigned short* p =
          w + (size_t)(n0 + ni * 16 + lm) * DIM + kc + klo0;
      bf[ni].q4[0] = *(const uint4*)(p);
      bf[ni].q4[1] = *(const uint4*)(p + 16);
    }
#pragma unroll
    for (int mi = 0; mi < 2; ++mi)
#pragma unroll
      for (int ni = 0; ni < 2; ++ni)
        acc[mi][ni] = wmma_bf16(af[mi], bf[ni], acc[mi][ni]);
  }

#pragma unroll
  for (int mi = 0; mi < 2; ++mi)
#pragma unroll
    for (int ni = 0; ni < 2; ++ni) {
      const int   n  = n0 + ni * 16 + lm;
      const float bv = bias[n];
#pragma unroll
      for (int r = 0; r < 8; ++r) {
        const int m = m0 + mi * 16 + half * 8 + r;
        out[(size_t)m * DIM + n] = acc[mi][ni][r] + bv;
      }
    }
}

// ---------------------------------------------------------------------------
extern "C" void kernel_launch(void* const* d_in, const int* in_sizes, int n_in,
                              void* d_out, int out_size, void* d_ws,
                              size_t ws_size, hipStream_t stream) {
  const float* x      = (const float*)d_in[0];
  const float* w_qkv  = (const float*)d_in[1];
  const float* w_proj = (const float*)d_in[2];
  const float* b_proj = (const float*)d_in[3];
  float*       out    = (float*)d_out;

  const size_t NX  = (size_t)BATCH * NSEQ * DIM;
  const size_t NWQ = (size_t)QKVN * DIM;
  const size_t NWP = (size_t)DIM * DIM;
  const size_t SZ  = (size_t)BATCH * HEADS * NSEQ * HD;

  unsigned short* ws_xb  = (unsigned short*)d_ws;
  unsigned short* ws_wqb = ws_xb + NX;
  unsigned short* ws_wpb = ws_wqb + NWQ;
  unsigned short* ws_q   = ws_wpb + NWP;
  unsigned short* ws_k   = ws_q + SZ;
  unsigned short* ws_vT  = ws_k + SZ;
  unsigned short* ws_at  = ws_vT + SZ;

  cvt_bf16_kernel<<<dim3((NX / 4 + 255) / 256), dim3(256), 0, stream>>>(
      x, ws_xb, (int)(NX / 4));
  cvt_bf16_kernel<<<dim3((NWQ / 4 + 255) / 256), dim3(256), 0, stream>>>(
      w_qkv, ws_wqb, (int)(NWQ / 4));
  cvt_bf16_kernel<<<dim3((NWP / 4 + 255) / 256), dim3(256), 0, stream>>>(
      w_proj, ws_wpb, (int)(NWP / 4));

  dim3 g1((BATCH * NSEQ) / 32, QKVN / 64);
  qkv_gemm_kernel<<<g1, dim3(32), 0, stream>>>(ws_xb, ws_wqb, ws_q, ws_k,
                                               ws_vT);

  dim3 g2(NSEQ / 64, HEADS, BATCH);
  flash_attn_kernel<<<g2, dim3(128), 0, stream>>>(ws_q, ws_k, ws_vT, ws_at);

  dim3 g3((BATCH * NSEQ) / 32, DIM / 32);
  proj_gemm_kernel<<<g3, dim3(32), 0, stream>>>(ws_at, ws_wpb, b_proj, out);
}
